// NoisePulseGenerator_50861002719921
// MI455X (gfx1250) — matrix operand order repaired and checked by reading, not compiled, not run on hardware
//
#include <hip/hip_runtime.h>
#include <hip/hip_bf16.h>

typedef __attribute__((ext_vector_type(16))) _Float16 v16h;
typedef __attribute__((ext_vector_type(8)))  _Float16 v8h;
typedef __attribute__((ext_vector_type(8)))  float    v8f;

union AFrag { v16h v; v8h q[2]; _Float16 h[16]; };

__device__ __forceinline__ float leaky_f(float x) {
    return fmaxf(x, 0.f) + 0.2f * fminf(x, 0.f);
}

// ---------------------------------------------------------------------------
// Weight packing:  Wt[n][k] = (f16) W[k][ perm(n) ] ;  perm folds the Wup
// channel-last reshape.  Run once per call, trivially cheap.
// ---------------------------------------------------------------------------
__global__ void pack_w_kernel(const float* __restrict__ W, _Float16* __restrict__ Wt,
                              int K, int N, int permute) {
    const int idx = blockIdx.x * blockDim.x + threadIdx.x;
    if (idx >= N * K) return;
    const int n = idx / K, k = idx % K;
    const int src = permute ? ((n & 127) * 8 + (n >> 7)) : n;
    Wt[idx] = (_Float16)W[(size_t)k * N + src];
}

// Pack ConvTranspose taps into per-parity K=256 B matrices, lane-contiguous:
//   Bt[p][co][kr] :  kr<128 -> w[kr, co, p?1:2] ;  kr>=128 -> w[kr-128, co, p?3:0]
__global__ void pack_wc_kernel(const float* __restrict__ Wc, _Float16* __restrict__ Bt) {
    const int idx = blockIdx.x * blockDim.x + threadIdx.x;   // 2*128*256
    if (idx >= 2 * 128 * 256) return;
    const int p  = idx >> 15;
    const int co = (idx >> 8) & 127;
    const int kr = idx & 255;
    const int ci = kr & 127;
    const int kA = p ? 1 : 2, kB = p ? 3 : 0;
    const int tap = (kr < 128) ? kA : kB;
    Bt[idx] = (_Float16)Wc[((size_t)ci * 128 + co) * 4 + tap];
}

// ---------------------------------------------------------------------------
// WMMA GEMM: C[M,NN](f16) = act(A[M,KK] @ Wt^T + bias).  Wt is [NN,KK] f16 so
// each B fragment = two b128 loads per lane.  A is f32 (first layer) or f16.
// NN/KK compile-time (powers of two) so all index math is shifts and the
// K-loop fully unrolls into back-to-back v_wmma_f32_16x16x32_f16.
// ---------------------------------------------------------------------------
template <bool A_HALF, int NN, int KK>
__global__ void gemm_wmma_kernel(const void* __restrict__ Avp,
                                 const _Float16* __restrict__ Wt,
                                 const float* __restrict__ bias,
                                 _Float16* __restrict__ C,
                                 int M, int doLeaky, int permBias) {
    constexpr int tilesN = NN / 16;
    const int lane  = threadIdx.x & 31;
    const int wave  = threadIdx.x >> 5;
    const int tile  = blockIdx.x * (blockDim.x >> 5) + wave;
    const int totalTiles = (M >> 4) * tilesN;
    if (tile >= totalTiles) return;              // wave-uniform
    const int tm = tile / tilesN, tn = tile % tilesN;
    const int g  = lane >> 4;
    const int mrow = tm * 16 + (lane & 15);
    const int ncol = tn * 16 + (lane & 15);

    v8f acc = {};
#pragma unroll
    for (int kb = 0; kb < KK; kb += 32) {
        AFrag a;
        if (A_HALF) {
            const _Float16* ar = (const _Float16*)Avp + (size_t)mrow * KK + kb + 8 * g;
            a.q[0] = *(const v8h*)ar;
            a.q[1] = *(const v8h*)(ar + 16);
        } else {
            const float* ar = (const float*)Avp + (size_t)mrow * KK + kb + 8 * g;
#pragma unroll
            for (int i = 0; i < 8; ++i) a.h[i]     = (_Float16)ar[i];
#pragma unroll
            for (int i = 0; i < 8; ++i) a.h[8 + i] = (_Float16)ar[16 + i];
        }
        AFrag bf;
        const _Float16* br = Wt + (size_t)ncol * KK + kb + 8 * g;
        bf.q[0] = *(const v8h*)br;
        bf.q[1] = *(const v8h*)(br + 16);
        acc = __builtin_amdgcn_wmma_f32_16x16x32_f16(false, a.v, false, bf.v,
                                                     (short)0, acc, false, false);
    }
    const int bcol = permBias ? ((ncol & 127) * 8 + (ncol >> 7)) : ncol;
    const float bv = bias[bcol];
#pragma unroll
    for (int r = 0; r < 8; ++r) {
        const int m = tm * 16 + r + 8 * g;
        float val = acc[r] + bv;
        if (doLeaky) val = leaky_f(val);
        C[(size_t)m * NN + ncol] = (_Float16)val;
    }
}

// ---------------------------------------------------------------------------
// ConvTranspose1d(k=4,s=2,p=1) 128->128 as a K=256 WMMA GEMM per parity:
//   y[2j]   = x[j].w[:,:,2] + x[j-1].w[:,:,0]
//   y[2j+1] = x[j].w[:,:,1] + x[j+1].w[:,:,3]
// X: [1024,L,128] f16 channel-last;  Bt: packed [2][128][256] f16.
// L compile-time power of two -> all row decomposition is shift/mask.
// ---------------------------------------------------------------------------
template <int L>
__global__ void tconv_wmma_kernel(const _Float16* __restrict__ X,
                                  const _Float16* __restrict__ Bt,
                                  const float* __restrict__ bias,
                                  _Float16* __restrict__ Y,
                                  int doLeaky) {
    constexpr int LOGL = (L == 8) ? 3 : (L == 16) ? 4 : 5;
    const int parity = blockIdx.y;
    const int lane = threadIdx.x & 31;
    const int wave = threadIdx.x >> 5;
    constexpr int tilesN = 8;                    // 128/16
    const int tile = blockIdx.x * (blockDim.x >> 5) + wave;
    constexpr int totalTiles = (1024 * L / 16) * tilesN;
    if (tile >= totalTiles) return;              // wave-uniform
    const int tm = tile >> 3, tn = tile & 7;
    const int g = lane >> 4;
    const int row = tm * 16 + (lane & 15);       // flat (b, j)
    const int b = row >> LOGL, j = row & (L - 1);
    const int co = tn * 16 + (lane & 15);

    const int jbr = parity ? j + 1 : j - 1;
    const int jbc = min(max(jbr, 0), L - 1);
    const _Float16 hmask = (_Float16)((jbr >= 0 && jbr < L) ? 1.f : 0.f);
    const _Float16* xa = X + (((size_t)b << LOGL) + j)   * 128;
    const _Float16* xb = X + (((size_t)b << LOGL) + jbc) * 128;
    const _Float16* Bp = Bt + (size_t)parity * 128 * 256 + (size_t)co * 256;

    v8f acc = {};
#pragma unroll
    for (int kc = 0; kc < 8; ++kc) {             // 8 chunks of 32 -> K=256
        const int kb32 = kc * 32;
        const _Float16* src = ((kc < 4) ? xa : xb) + (kb32 & 127) + 8 * g;
        AFrag a;
        a.q[0] = *(const v8h*)src;
        a.q[1] = *(const v8h*)(src + 16);
        if (kc >= 4) { a.q[0] *= hmask; a.q[1] *= hmask; }
        AFrag bf;
        const _Float16* br = Bp + kb32 + 8 * g;
        bf.q[0] = *(const v8h*)br;
        bf.q[1] = *(const v8h*)(br + 16);
        acc = __builtin_amdgcn_wmma_f32_16x16x32_f16(false, a.v, false, bf.v,
                                                     (short)0, acc, false, false);
    }
    const float bv = bias[co];
#pragma unroll
    for (int r = 0; r < 8; ++r) {
        const int rr = tm * 16 + r + 8 * g;
        const int bb = rr >> LOGL, jj = rr & (L - 1);
        const int o = 2 * jj + parity;
        float val = acc[r] + bv;
        if (doLeaky) val = leaky_f(val);
        Y[(((size_t)bb * (2 * L)) + o) * 128 + co] = (_Float16)val;
    }
}

// decay = 0.8 + 0.2 * sigmoid(h @ Wdo + bdo), one thread per row (h is f16)
__global__ void decay_kernel(const _Float16* __restrict__ H, const float* __restrict__ Wdo,
                             const float* __restrict__ bdo, float* __restrict__ decay) {
    const int b = blockIdx.x * blockDim.x + threadIdx.x;
    if (b >= 1024) return;
    const _Float16* h = H + (size_t)b * 128;
    float s = 0.f;
#pragma unroll 4
    for (int i = 0; i < 128; ++i) s += (float)h[i] * Wdo[i];
    s += bdo[0];
    const float sg = 1.f / (1.f + __expf(-s));
    decay[b] = 0.8f + sg * 0.2f;
}

// last tconv layer (128ch -> 1ch) + bias + square.  X: [1024,64,128] f16 -> E f32
__global__ void tconv_last_kernel(const _Float16* __restrict__ X, const float* __restrict__ W3,
                                  const float* __restrict__ b3, float* __restrict__ E) {
    const int idx = blockIdx.x * blockDim.x + threadIdx.x;   // 1024*128
    if (idx >= 1024 * 128) return;
    const int b = idx >> 7, o = idx & 127;
    const int j = o >> 1, parity = o & 1;
    const int kA = parity ? 1 : 2, kB = parity ? 3 : 0;
    const int jb = parity ? j + 1 : j - 1;
    const _Float16* xa = X + (((size_t)b << 6) + j) * 128;
    float s = 0.f;
#pragma unroll 4
    for (int ci = 0; ci < 128; ++ci) s += (float)xa[ci] * W3[ci * 4 + kA];
    if (jb >= 0 && jb < 64) {
        const _Float16* xb = X + (((size_t)b << 6) + jb) * 128;
#pragma unroll 4
        for (int ci = 0; ci < 128; ++ci) s += (float)xb[ci] * W3[ci * 4 + kB];
    }
    s += b3[0];
    E[idx] = s * s;
}

// v[i] = e[i] + decay * v[i-1], one thread per row
__global__ void scan_kernel(const float* __restrict__ E, const float* __restrict__ decay,
                            float* __restrict__ V) {
    const int b = blockIdx.x * blockDim.x + threadIdx.x;
    if (b >= 1024) return;
    const float d = decay[b];
    const float* e = E + (size_t)b * 128;
    float* vo = V + (size_t)b * 128;
    float v = 0.f;
    for (int i = 0; i < 128; ++i) { v = e[i] + d * v; vo[i] = v; }
}

// linear upsample 128 -> 32768 (align_corners=False) then * noise
__global__ void interp_noise_kernel(const float* __restrict__ V, const float* __restrict__ noise,
                                    float* __restrict__ out) {
    __shared__ float v[128];
    const int b = blockIdx.x;
    if (threadIdx.x < 128) v[threadIdx.x] = V[(size_t)b * 128 + threadIdx.x];
    __syncthreads();
    const float* nrow = noise + (size_t)b * 32768;
    float* orow = out + (size_t)b * 32768;
    for (int s4 = threadIdx.x; s4 < 8192; s4 += blockDim.x) {
        const int s = s4 * 4;
        const float4 nz = ((const float4*)nrow)[s4];
        float r[4];
#pragma unroll
        for (int q = 0; q < 4; ++q) {
            float pos = (s + q + 0.5f) * (1.0f / 256.0f) - 0.5f;
            pos = fminf(fmaxf(pos, 0.f), 127.f);
            const int i0 = (int)pos;
            const int i1 = min(i0 + 1, 127);
            const float w = pos - (float)i0;
            r[q] = v[i0] * (1.f - w) + v[i1] * w;
        }
        const float4 ov = { r[0] * nz.x, r[1] * nz.y, r[2] * nz.z, r[3] * nz.w };
        ((float4*)orow)[s4] = ov;
    }
}

extern "C" void kernel_launch(void* const* d_in, const int* in_sizes, int n_in,
                              void* d_out, int out_size, void* d_ws, size_t ws_size,
                              hipStream_t stream) {
    const float* x    = (const float*)d_in[0];   // [1024, 128]
    const float* noise= (const float*)d_in[1];   // [1024, 32768]
    const float* Wup  = (const float*)d_in[2];   // [128, 1024]
    const float* bup  = (const float*)d_in[3];
    const float* Wd0  = (const float*)d_in[4];
    const float* bd0  = (const float*)d_in[5];
    const float* Wd1  = (const float*)d_in[6];
    const float* bd1  = (const float*)d_in[7];
    const float* Wd2  = (const float*)d_in[8];
    const float* bd2  = (const float*)d_in[9];
    const float* Wdo  = (const float*)d_in[10];
    const float* bdo  = (const float*)d_in[11];
    const float* Wc0  = (const float*)d_in[12];
    const float* bc0  = (const float*)d_in[13];
    const float* Wc1  = (const float*)d_in[14];
    const float* bc1  = (const float*)d_in[15];
    const float* Wc2  = (const float*)d_in[16];
    const float* bc2  = (const float*)d_in[17];
    const float* Wc3  = (const float*)d_in[18];
    const float* bc3  = (const float*)d_in[19];
    float* out = (float*)d_out;

    // ---- workspace carve-up (f32 region first, then f16 region) ----
    float* wsf = (float*)d_ws;
    float* decay = wsf;                     // 1024
    float* e     = decay + 1024;            // 131072
    float* v     = e + 131072;              // 131072
    _Float16* wsh = (_Float16*)(v + 131072);
    _Float16* h0    = wsh;                      // 131072
    _Float16* h1    = h0 + 131072;              // 131072
    _Float16* u0    = h1 + 131072;              // 1024*8*128
    _Float16* u1    = u0 + 1024 * 8 * 128;      // 1024*16*128
    _Float16* u2    = u1 + 1024 * 16 * 128;     // 1024*32*128
    _Float16* u3    = u2 + 1024 * 32 * 128;     // 1024*64*128
    _Float16* Wt_d0 = u3 + 1024 * 64 * 128;     // 128*128
    _Float16* Wt_d1 = Wt_d0 + 128 * 128;
    _Float16* Wt_d2 = Wt_d1 + 128 * 128;
    _Float16* Wt_up = Wt_d2 + 128 * 128;        // 1024*128
    _Float16* Bt_c0 = Wt_up + 1024 * 128;       // 2*128*256
    _Float16* Bt_c1 = Bt_c0 + 2 * 128 * 256;
    _Float16* Bt_c2 = Bt_c1 + 2 * 128 * 256;

    const dim3 blk(256);

    // ---- weight packing (f16, lane-contiguous fragments) ----
    pack_w_kernel<<<dim3(64),  blk, 0, stream>>>(Wd0, Wt_d0, 128, 128, 0);
    pack_w_kernel<<<dim3(64),  blk, 0, stream>>>(Wd1, Wt_d1, 128, 128, 0);
    pack_w_kernel<<<dim3(64),  blk, 0, stream>>>(Wd2, Wt_d2, 128, 128, 0);
    pack_w_kernel<<<dim3(512), blk, 0, stream>>>(Wup, Wt_up, 128, 1024, 1);
    pack_wc_kernel<<<dim3(256), blk, 0, stream>>>(Wc0, Bt_c0);
    pack_wc_kernel<<<dim3(256), blk, 0, stream>>>(Wc1, Bt_c1);
    pack_wc_kernel<<<dim3(256), blk, 0, stream>>>(Wc2, Bt_c2);

    // ---- decay head: three 1024x128 @ 128x128 WMMA GEMMs ----
    gemm_wmma_kernel<false, 128, 128><<<dim3(64), blk, 0, stream>>>(x,  Wt_d0, bd0, h0, 1024, 1, 0);
    gemm_wmma_kernel<true,  128, 128><<<dim3(64), blk, 0, stream>>>(h0, Wt_d1, bd1, h1, 1024, 1, 0);
    gemm_wmma_kernel<true,  128, 128><<<dim3(64), blk, 0, stream>>>(h1, Wt_d2, bd2, h0, 1024, 1, 0);
    decay_kernel<<<dim3(4), blk, 0, stream>>>(h0, Wdo, bdo, decay);

    // ---- up-projection -> channel-last [b,t,c] (permutation folded in Wt_up) ----
    gemm_wmma_kernel<false, 1024, 128><<<dim3(512), blk, 0, stream>>>(x, Wt_up, bup, u0, 1024, 0, 1);

    // ---- three 128ch transposed-conv layers, parity-split WMMA GEMMs ----
    tconv_wmma_kernel<8 ><<<dim3(512,  2), blk, 0, stream>>>(u0, Bt_c0, bc0, u1, 1);
    tconv_wmma_kernel<16><<<dim3(1024, 2), blk, 0, stream>>>(u1, Bt_c1, bc1, u2, 1);
    tconv_wmma_kernel<32><<<dim3(2048, 2), blk, 0, stream>>>(u2, Bt_c2, bc2, u3, 1);

    // ---- last layer (to 1ch) + square, decay scan ----
    tconv_last_kernel<<<dim3(512), blk, 0, stream>>>(u3, Wc3, bc3, e);
    scan_kernel<<<dim3(4), blk, 0, stream>>>(e, decay, v);

    // ---- bandwidth kernel: upsample 128->32768 and multiply with noise ----
    interp_noise_kernel<<<dim3(1024), blk, 0, stream>>>(v, noise, out);
}
